// GCN_678604833480
// MI455X (gfx1250) — matrix operand order repaired and checked by reading, not compiled, not run on hardware
//
#include <hip/hip_runtime.h>

// ---------------- problem constants ----------------
#define NB   8
#define NL   2304            // nodes per batch (48*48)
#define BL   18432           // NB*NL
#define C0   200
#define INUM 1024
#define OC   64
#define EPSI 1e-5f
#define SLOPE 0.01f

typedef __attribute__((ext_vector_type(4)))  float   f4;
typedef __attribute__((ext_vector_type(8)))  float   v8f;
typedef __attribute__((ext_vector_type(16))) __bf16  v16bf;
typedef int v4i __attribute__((vector_size(16)));

union ABFrag { f4 q[2]; v16bf v; };
union AccU   { v8f v; float f[8]; };

// -------- CDNA5 async global->LDS path (probe via __has_builtin) --------
#if defined(__has_builtin)
#if __has_builtin(__builtin_amdgcn_global_load_async_to_lds_b128)
#define HAVE_ASYNC_LDS 1
#endif
#if __has_builtin(__builtin_amdgcn_s_wait_asynccnt)
#define HAVE_WAIT_ASYNC 1
#endif
#endif

#if HAVE_ASYNC_LDS
__device__ __forceinline__ void async_cp16(const __bf16* g, __bf16* l) {
  __builtin_amdgcn_global_load_async_to_lds_b128(
      (__attribute__((address_space(1))) v4i*)(void*)g,
      (__attribute__((address_space(3))) v4i*)(void*)l, 0, 0);
}
__device__ __forceinline__ void wait_async_all() {
#if HAVE_WAIT_ASYNC
  __builtin_amdgcn_s_wait_asynccnt(0);
#else
  asm volatile("s_wait_asynccnt 0" ::: "memory");
#endif
}
#endif

// ---------------- K1: deg + A -> bf16 ----------------
__global__ __launch_bounds__(256) void k_deg(const float* __restrict__ A,
                                             __bf16* __restrict__ A16,
                                             float* __restrict__ rs,
                                             float* __restrict__ rsinv) {
  __shared__ float red[256];
  size_t r = blockIdx.x;                 // 0..BL-1
  int t = threadIdx.x;
  const float* row = A + r * NL;
  __bf16* row16 = A16 + r * NL;
  float s = 0.f;
  #pragma unroll
  for (int i = 0; i < 9; i++) {          // 2304 = 9*256
    int j = t + i * 256;
    float v = row[j];
    row16[j] = (__bf16)v;
    s += v;
  }
  red[t] = s; __syncthreads();
  for (int off = 128; off > 0; off >>= 1) {
    if (t < off) red[t] += red[t + off];
    __syncthreads();
  }
  if (t == 0) { float d = red[0]; rs[r] = sqrtf(d); rsinv[r] = rsqrtf(d); }
}

// ---------------- K2: per-channel column sums ----------------
__global__ __launch_bounds__(256) void k_stats(const float* __restrict__ H, int ci,
                                               float* __restrict__ ssum,
                                               float* __restrict__ ssq) {
  int t = threadIdx.x;
  if (t >= ci) return;
  size_t r0 = (size_t)blockIdx.x * 144;  // 128 blocks * 144 rows = 18432
  float s = 0.f, q = 0.f;
  for (int i = 0; i < 144; i++) {
    float v = H[(r0 + i) * ci + t];
    s += v; q += v * v;
  }
  atomicAdd(&ssum[t], s);
  atomicAdd(&ssq[t], q);
}

// ---------------- K3: fold BN into W/c ----------------
__global__ __launch_bounds__(256) void k_finalize(const float* __restrict__ ssum,
                                                  const float* __restrict__ ssq,
                                                  const float* __restrict__ g,
                                                  const float* __restrict__ bvec,
                                                  const float* __restrict__ W,
                                                  const float* __restrict__ cbias,
                                                  int ci, int co,
                                                  float* __restrict__ W2,
                                                  float* __restrict__ c2) {
  __shared__ float sSc[256], sSh[256];
  int t = threadIdx.x;
  if (t < ci) {
    float mu  = ssum[t] * (1.f / (float)BL);
    float var = ssq[t]  * (1.f / (float)BL) - mu * mu;
    float sc  = g[t] * rsqrtf(var + EPSI);
    sSc[t] = sc;
    sSh[t] = bvec[t] - mu * sc;
  }
  __syncthreads();
  for (int idx = t; idx < ci * co; idx += 256)
    W2[idx] = sSc[idx / co] * W[idx];
  if (t < co) {
    float s = cbias[t];
    for (int k = 0; k < ci; k++) s += sSh[k] * W[k * co + t];
    c2[t] = s;
  }
}

// ---------------- K4: M = H@W2 + c2 ; MsT = bf16(M*rsinv) transposed ----------------
template <int CO>
__global__ __launch_bounds__(256) void k_mgemm(const float* __restrict__ Hin, int ci,
                                               const float* __restrict__ W2,
                                               const float* __restrict__ c2,
                                               const float* __restrict__ rsinv,
                                               float* __restrict__ Mbuf,
                                               __bf16* __restrict__ MsT) {
  constexpr int GW = CO / 4;
  __shared__ float sW[64 * CO];
  int t = threadIdx.x;
  int rl = t >> 2;                 // 0..63 local row
  int cg = t & 3;                  // column group
  size_t row = (size_t)blockIdx.x * 64 + rl;
  float acc[GW];
  #pragma unroll
  for (int j = 0; j < GW; j++) acc[j] = 0.f;
  for (int kb = 0; kb < ci; kb += 64) {
    int kend = min(64, ci - kb);
    __syncthreads();
    for (int idx = t; idx < kend * CO; idx += 256) sW[idx] = W2[(size_t)kb * CO + idx];
    __syncthreads();
    for (int k = 0; k < kend; k++) {
      float h = Hin[row * ci + kb + k];
      #pragma unroll
      for (int j = 0; j < GW; j++) acc[j] += h * sW[k * CO + cg * GW + j];
    }
  }
  int bb = (int)(row / NL);
  int il = (int)(row - (size_t)bb * NL);
  float ri = rsinv[row];
  #pragma unroll
  for (int j = 0; j < GW; j++) {
    int c = cg * GW + j;
    float val = acc[j] + c2[c];
    Mbuf[row * CO + c] = val;
    MsT[((size_t)bb * CO + c) * NL + il] = (__bf16)(val * ri);
  }
}

// ---------------- K5 compute step: 1 A-frag x TN B-frags of WMMA ----------------
template <int CO>
__device__ __forceinline__ void compute_step(const __bf16* lAb, const __bf16* lBb,
                                             AccU* acc, int wm, int wn, int l15, int lh) {
  constexpr int TN = CO / 32;
  // A fragment: ISA 16-bit A 16x32 layout (lane<16: K0-7,16-23; lane>=16: K8-15,24-31)
  ABFrag af;
  const __bf16* pa = &lAb[(wm * 16 + l15) * 40 + lh * 8];
  af.q[0] = *(const f4*)pa;
  af.q[1] = *(const f4*)(pa + 16);
  // load ALL B fragments first so ds_loads pipeline, then chain WMMAs
  ABFrag bfr[TN];
  #pragma unroll
  for (int tn = 0; tn < TN; tn++) {
    const __bf16* pb = &lBb[(wn * (CO / 2) + tn * 16 + l15) * 40 + lh * 8];
    bfr[tn].q[0] = *(const f4*)pb;
    bfr[tn].q[1] = *(const f4*)(pb + 16);
  }
  #pragma unroll
  for (int tn = 0; tn < TN; tn++)
    acc[tn].v = __builtin_amdgcn_wmma_f32_16x16x32_bf16(
        false, af.v, false, bfr[tn].v, (short)0, acc[tn].v, false, false);
}

// ---------------- K5: P = A @ Ms via WMMA bf16 (double-buffered); out = leaky(M - rs*P) ----------------
template <int CO>
__global__ __launch_bounds__(256) void k_agemm(const __bf16* __restrict__ A16,
                                               const __bf16* __restrict__ MsT,
                                               const float* __restrict__ Mbuf,
                                               const float* __restrict__ rs,
                                               float* __restrict__ Hout) {
  constexpr int TN  = CO / 32;     // 16-col tiles per wave
  constexpr int LDA = 40;          // bf16 per LDS row (80B, padded)
  constexpr int ASZ = 64 * LDA;
  constexpr int BSZ = CO * LDA;
  __shared__ __bf16 lA[2 * ASZ];
  __shared__ __bf16 lB[2 * BSZ];
  const int b    = blockIdx.y;
  const int i0   = blockIdx.x * 64;
  const int t    = threadIdx.x;
  const int lane = t & 31;
  const int wave = t >> 5;
  const int wm   = wave >> 1;      // 0..3 : 16-row slab
  const int wn   = wave & 1;       // 0..1 : CO/2 column slab
  const int l15  = lane & 15;
  const int lh   = lane >> 4;      // 0/1

  AccU acc[TN];
  #pragma unroll
  for (int i = 0; i < TN; i++) acc[i].v = (v8f){0, 0, 0, 0, 0, 0, 0, 0};

  // tile-copy mapping: 16B chunk per thread per 64-row slab
  const int arow = t >> 2, ach = t & 3;
  const __bf16* gA = A16 + ((size_t)b * NL + i0 + arow) * NL + ach * 8;
  const __bf16* gB = MsT + ((size_t)b * CO + arow) * NL + ach * 8;
  const int dOff = arow * LDA + ach * 8;

  int cur = 0;
#if HAVE_ASYNC_LDS
  // -------- async global->LDS double-buffered pipeline --------
  auto issue = [&](int kk, int nb) {
    async_cp16(gA + kk, &lA[nb * ASZ + dOff]);
    #pragma unroll
    for (int r = 0; r < CO / 64; r++)
      async_cp16(gB + (size_t)r * 64 * NL + kk, &lB[nb * BSZ + r * 64 * LDA + dOff]);
  };
  issue(0, 0);
  wait_async_all();
  __syncthreads();
  for (int kk = 0; kk < NL; kk += 32) {
    if (kk + 32 < NL) issue(kk + 32, cur ^ 1);
    compute_step<CO>(&lA[cur * ASZ], &lB[cur * BSZ], acc, wm, wn, l15, lh);
    wait_async_all();
    __syncthreads();
    cur ^= 1;
  }
#else
  // -------- register-staged double-buffered pipeline --------
  f4 ra, rb[CO / 64];
  auto loadr = [&](int kk) {
    ra = *(const f4*)(gA + kk);
    #pragma unroll
    for (int r = 0; r < CO / 64; r++) rb[r] = *(const f4*)(gB + (size_t)r * 64 * NL + kk);
  };
  auto storer = [&](int nb) {
    *(f4*)&lA[nb * ASZ + dOff] = ra;
    #pragma unroll
    for (int r = 0; r < CO / 64; r++) *(f4*)&lB[nb * BSZ + r * 64 * LDA + dOff] = rb[r];
  };
  loadr(0); storer(0);
  __syncthreads();
  for (int kk = 0; kk < NL; kk += 32) {
    bool more = (kk + 32) < NL;
    if (more) loadr(kk + 32);
    compute_step<CO>(&lA[cur * ASZ], &lB[cur * BSZ], acc, wm, wn, l15, lh);
    if (more) storer(cur ^ 1);
    __syncthreads();
    cur ^= 1;
  }
#endif

  // epilogue: out = leaky(M - rs_i * P)
  #pragma unroll
  for (int tn = 0; tn < TN; tn++) {
    #pragma unroll
    for (int v = 0; v < 8; v++) {
      int i = i0 + wm * 16 + v + lh * 8;
      int c = wn * (CO / 2) + tn * 16 + l15;
      size_t grow = (size_t)b * NL + i;
      size_t o = grow * CO + c;
      float val = Mbuf[o] - rs[grow] * acc[tn].f[v];
      Hout[o] = val >= 0.f ? val : SLOPE * val;
    }
  }
}

// ---------------- K6: gather + per-(b,c) mean/max ----------------
__global__ __launch_bounds__(128) void k_gather(const float* __restrict__ Hf,
                                                const int* __restrict__ idx,
                                                float* __restrict__ G,
                                                float* __restrict__ avg,
                                                float* __restrict__ mx) {
  int b = blockIdx.x >> 6, c = blockIdx.x & 63;
  int t = threadIdx.x;
  __shared__ float rsum[128], rmax[128];
  float s = 0.f, m = -3.4e38f;
  #pragma unroll
  for (int i = 0; i < 8; i++) {
    int tt = t + i * 128;
    int node = idx[b * INUM + tt];
    float v = Hf[((size_t)b * NL + node) * OC + c];
    G[((size_t)b * OC + c) * INUM + tt] = v;
    s += v; m = fmaxf(m, v);
  }
  rsum[t] = s; rmax[t] = m; __syncthreads();
  for (int off = 64; off > 0; off >>= 1) {
    if (t < off) { rsum[t] += rsum[t + off]; rmax[t] = fmaxf(rmax[t], rmax[t + off]); }
    __syncthreads();
  }
  if (t == 0) { avg[blockIdx.x] = rsum[0] * (1.f / INUM); mx[blockIdx.x] = rmax[0]; }
}

// ---------------- K7: channel-attention MLP ----------------
__global__ __launch_bounds__(64) void k_chatt(const float* __restrict__ avg,
                                              const float* __restrict__ mx,
                                              const float* __restrict__ fc1,
                                              const float* __restrict__ fc2,
                                              float* __restrict__ ca) {
  int b = blockIdx.x, t = threadIdx.x;
  __shared__ float h1a[4], h1m[4];
  if (t < 8) {
    int o = t & 3;
    const float* v = (t < 4) ? (avg + b * 64) : (mx + b * 64);
    float s = 0.f;
    for (int i = 0; i < 64; i++) s += fc1[o * 64 + i] * v[i];
    s = fmaxf(s, 0.f);
    if (t < 4) h1a[o] = s; else h1m[o] = s;
  }
  __syncthreads();
  float s = 0.f;
  #pragma unroll
  for (int o = 0; o < 4; o++) s += fc2[t * 4 + o] * (h1a[o] + h1m[o]);
  ca[b * 64 + t] = 1.f / (1.f + expf(-s));
}

// ---------------- K8: apply ca; spatial mean/max planes ----------------
__global__ __launch_bounds__(256) void k_spat(const float* __restrict__ ca,
                                              float* __restrict__ G,
                                              float* __restrict__ sm,
                                              float* __restrict__ sx) {
  int gt = blockIdx.x * 256 + threadIdx.x;   // < NB*INUM
  int b = gt >> 10, t = gt & 1023;
  float s = 0.f, m = -3.4e38f;
  for (int c = 0; c < 64; c++) {
    size_t o = ((size_t)b * OC + c) * INUM + t;
    float v = ca[b * 64 + c] * G[o];
    G[o] = v;
    s += v; m = fmaxf(m, v);
  }
  sm[gt] = s * (1.f / 64.f); sx[gt] = m;
}

// ---------------- K9a: 3-tap conv + sigmoid ----------------
__global__ __launch_bounds__(256) void k_convsig(const float* __restrict__ sm,
                                                 const float* __restrict__ sx,
                                                 const float* __restrict__ saw,
                                                 float* __restrict__ sig) {
  int gt = blockIdx.x * 256 + threadIdx.x;
  int b = gt >> 10, t = gt & 1023;
  float s = 0.f;
  #pragma unroll
  for (int kh = 0; kh < 3; kh++) {
    int tt = t + kh - 1;
    if (tt >= 0 && tt < INUM)
      s += saw[kh * 3 + 1] * sm[b * INUM + tt] + saw[9 + kh * 3 + 1] * sx[b * INUM + tt];
  }
  sig[gt] = 1.f / (1.f + expf(-s));
}

// ---------------- K9b: apply spatial att + BN stats ----------------
__global__ __launch_bounds__(256) void k_apply(const float* __restrict__ sig,
                                               float* __restrict__ G,
                                               float* __restrict__ bnsum,
                                               float* __restrict__ bnsq) {
  int b = blockIdx.x >> 6, c = blockIdx.x & 63;
  int t = threadIdx.x;
  __shared__ float r1[256], r2[256];
  float s = 0.f, q = 0.f;
  #pragma unroll
  for (int i = 0; i < 4; i++) {
    int tt = t + i * 256;
    size_t o = ((size_t)b * OC + c) * INUM + tt;
    float v = sig[b * INUM + tt] * G[o];
    G[o] = v;
    s += v; q += v * v;
  }
  r1[t] = s; r2[t] = q; __syncthreads();
  for (int off = 128; off > 0; off >>= 1) {
    if (t < off) { r1[t] += r1[t + off]; r2[t] += r2[t + off]; }
    __syncthreads();
  }
  if (t == 0) { atomicAdd(&bnsum[c], r1[0]); atomicAdd(&bnsq[c], r2[0]); }
}

// ---------------- K10: final BN ----------------
__global__ __launch_bounds__(256) void k_bnout(const float* __restrict__ G,
                                               const float* __restrict__ bnsum,
                                               const float* __restrict__ bnsq,
                                               const float* __restrict__ bng,
                                               const float* __restrict__ bnb,
                                               float* __restrict__ out) {
  int gi = blockIdx.x * 256 + threadIdx.x;   // < NB*OC*INUM
  int c = (gi >> 10) & 63;
  float mu  = bnsum[c] * (1.f / 8192.f);
  float var = bnsq[c]  * (1.f / 8192.f) - mu * mu;
  out[gi] = (G[gi] - mu) * rsqrtf(var + EPSI) * bng[c] + bnb[c];
}

// ---------------- host ----------------
extern "C" void kernel_launch(void* const* d_in, const int* in_sizes, int n_in,
                              void* d_out, int out_size, void* d_ws, size_t ws_size,
                              hipStream_t stream) {
  const float* x    = (const float*)d_in[0];
  const float* A    = (const float*)d_in[1];
  const int*   idx  = (const int*)d_in[2];
  const float* fc1  = (const float*)d_in[19];
  const float* fc2  = (const float*)d_in[20];
  const float* saw  = (const float*)d_in[21];
  const float* bng  = (const float*)d_in[22];
  const float* bnb  = (const float*)d_in[23];

  char* ws = (char*)d_ws;
  size_t off = 0;
  auto alloc = [&](size_t bytes) -> char* {
    char* p = ws + off;
    off += (bytes + 255) & ~(size_t)255;
    return p;
  };
  __bf16* A16   = (__bf16*)alloc((size_t)BL * NL * 2);           // 85 MB, L2-resident
  float*  rs    = (float*)alloc((size_t)BL * 4);
  float*  rsinv = (float*)alloc((size_t)BL * 4);
  float*  Hb0   = (float*)alloc((size_t)BL * 128 * 4);
  float*  Hb1   = (float*)alloc((size_t)BL * 128 * 4);
  float*  Mbuf  = (float*)alloc((size_t)BL * 128 * 4);
  __bf16* MsT   = (__bf16*)alloc((size_t)NB * 128 * NL * 2);
  float*  ssum  = (float*)alloc(2 * 256 * 4);                    // ssum+ssq contiguous
  float*  ssq   = ssum + 256;
  float*  W2    = (float*)alloc(256 * 128 * 4);
  float*  c2    = (float*)alloc(128 * 4);
  float*  G     = (float*)alloc((size_t)NB * OC * INUM * 4);
  float*  avg   = (float*)alloc(NB * 64 * 4);
  float*  mx    = (float*)alloc(NB * 64 * 4);
  float*  ca    = (float*)alloc(NB * 64 * 4);
  float*  sm    = (float*)alloc(NB * INUM * 4);
  float*  sx    = (float*)alloc(NB * INUM * 4);
  float*  sig   = (float*)alloc(NB * INUM * 4);
  float*  bnsum = (float*)alloc(2 * 64 * 4);                     // bnsum+bnsq contiguous
  float*  bnsq  = bnsum + 64;

  // deg + A->bf16 (once; A_hat is layer-invariant)
  k_deg<<<BL, 256, 0, stream>>>(A, A16, rs, rsinv);

  const float* Hin[4]  = {x, Hb0, Hb1, Hb0};
  float*       Hout[4] = {Hb0, Hb1, Hb0, Hb1};
  const int    ciL[4]  = {C0, 128, 128, 128};
  const int    coL[4]  = {128, 128, 128, 64};

  for (int l = 0; l < 4; l++) {
    const float* g  = (const float*)d_in[3 + 4 * l + 0];
    const float* bv = (const float*)d_in[3 + 4 * l + 1];
    const float* W  = (const float*)d_in[3 + 4 * l + 2];
    const float* cb = (const float*)d_in[3 + 4 * l + 3];
    int ci = ciL[l], co = coL[l];
    (void)hipMemsetAsync(ssum, 0, 2 * 256 * 4, stream);
    k_stats<<<128, 256, 0, stream>>>(Hin[l], ci, ssum, ssq);
    k_finalize<<<1, 256, 0, stream>>>(ssum, ssq, g, bv, W, cb, ci, co, W2, c2);
    if (co == 128) {
      k_mgemm<128><<<BL / 64, 256, 0, stream>>>(Hin[l], ci, W2, c2, rsinv, Mbuf, MsT);
      k_agemm<128><<<dim3(NL / 64, NB), 256, 0, stream>>>(A16, MsT, Mbuf, rs, Hout[l]);
    } else {
      k_mgemm<64><<<BL / 64, 256, 0, stream>>>(Hin[l], ci, W2, c2, rsinv, Mbuf, MsT);
      k_agemm<64><<<dim3(NL / 64, NB), 256, 0, stream>>>(A16, MsT, Mbuf, rs, Hout[l]);
    }
  }

  // gather + attention + final BN
  k_gather<<<NB * 64, 128, 0, stream>>>(Hout[3], idx, G, avg, mx);
  k_chatt<<<NB, 64, 0, stream>>>(avg, mx, fc1, fc2, ca);
  k_spat<<<NB * INUM / 256, 256, 0, stream>>>(ca, G, sm, sx);
  k_convsig<<<NB * INUM / 256, 256, 0, stream>>>(sm, sx, saw, sig);
  (void)hipMemsetAsync(bnsum, 0, 2 * 64 * 4, stream);
  k_apply<<<NB * 64, 256, 0, stream>>>(sig, G, bnsum, bnsq);
  k_bnout<<<NB * OC * INUM / 256, 256, 0, stream>>>(G, bnsum, bnsq, bng, bnb, (float*)d_out);
}
